// GraphNet_66666482368869
// MI455X (gfx1250) — matrix operand order repaired and checked by reading, not compiled, not run on hardware
//
#include <hip/hip_runtime.h>
#include <hip/hip_fp16.h>

#define NN 50000
#define NE 800000
#define DD 64
#define ESZ 16
#define RR 3
#define MSGK 144
#define MSGP 160   // K padded to multiple of 32
#define HGD 128

typedef _Float16 f16;
typedef __attribute__((ext_vector_type(16))) _Float16 v16h;
typedef __attribute__((ext_vector_type(8)))  _Float16 v8h;
typedef __attribute__((ext_vector_type(8)))  float    v8f;

// ---- WMMA fragment helpers (contiguous k = 16*(lane/16)+i, same order in A and B) ----

// A: 16x32 f16 tile from LDS (row-major, row stride ldk, ldk multiple of 8).
// lane<16 : row m=lane,    K-offsets {0..15}
// lane>=16: row m=lane-16, K-offsets {16..31}
static __device__ __forceinline__ v16h load_frag_a(const f16* As, int ldk, int k0, int lane) {
    int m    = lane & 15;
    int base = (lane >> 4) << 4;           // 0 or 16
    const v8h* p = (const v8h*)(As + m * ldk + k0 + base);
    v8h lo = p[0];                          // K-offsets base..base+7
    v8h hi = p[1];                          // K-offsets base+8..base+15
    v16h a;
#pragma unroll
    for (int i = 0; i < 8; ++i) { a[i] = lo[i]; a[8 + i] = hi[i]; }
    return a;
}

// B: 32x16, B[k][n] = W[oBase+n][k] with W pre-converted f16, row-major [O][ldw],
// ldw multiple of 8 (zero-padded at conversion, so no bounds guard needed).
// lane holds column n=lane&15; lanes<16 hold K=k0..k0+15, lanes>=16 K=k0+16..k0+31.
static __device__ __forceinline__ v16h load_frag_b(const f16* __restrict__ W, int ldw,
                                                   int oBase, int k0, int lane) {
    int n  = lane & 15;
    int kh = k0 + ((lane >> 4) << 4);
    const v8h* p = (const v8h*)(W + (size_t)(oBase + n) * ldw + kh);
    v8h lo = p[0];
    v8h hi = p[1];
    v16h b;
#pragma unroll
    for (int i = 0; i < 8; ++i) { b[i] = lo[i]; b[8 + i] = hi[i]; }
    return b;
}

#define WMMA_F16(a, b, c) \
    __builtin_amdgcn_wmma_f32_16x16x32_f16(false, (a), false, (b), (short)0, (c), false, false)

static __device__ __forceinline__ float fast_sigmoid(float x) {
    return 1.0f / (1.0f + __expf(-x));
}

// ---- Convert f32 weights to f16 with optional K zero-padding ----
__global__ void cvt_pad_kernel(const float* __restrict__ src, f16* __restrict__ dst,
                               int rows, int ksrc, int kdst) {
    int idx = blockIdx.x * blockDim.x + threadIdx.x;
    if (idx >= rows * kdst) return;
    int r = idx / kdst, k = idx % kdst;
    dst[idx] = (k < ksrc) ? (f16)src[(size_t)r * ksrc + k] : (f16)0.0f;
}

// ---- Round-invariant edge aggregates: Ein/Eout [N,16], deg_in/deg_out [N] ----
__global__ void edge_const_kernel(const float* __restrict__ ev,
                                  const int* __restrict__ u, const int* __restrict__ v,
                                  float* __restrict__ Ein, float* __restrict__ Eout,
                                  float* __restrict__ deg_in, float* __restrict__ deg_out) {
    int e = blockIdx.x * blockDim.x + threadIdx.x;
    if (e >= NE) return;
    int ue = u[e], ve = v[e];
    atomicAdd(&deg_in[ve], 1.0f);
    atomicAdd(&deg_out[ue], 1.0f);
    const float* evr = ev + (size_t)e * ESZ;
#pragma unroll
    for (int c = 0; c < ESZ; ++c) {
        float x = evr[c];
        atomicAdd(&Ein[(size_t)ve * ESZ + c], x);
        atomicAdd(&Eout[(size_t)ue * ESZ + c], x);
    }
}

// ---- Per-round scatter: S_in[v]+=h[u], S_out[u]+=h[v] ----
// f16 accumulators via packed hardware atomics (GLOBAL_ATOMIC_PK_ADD_F16),
// all traffic L2-resident (S buffers are 6.4MB each, h is 12.8MB).
__global__ void scatter_kernel(const float* __restrict__ h,
                               const int* __restrict__ u, const int* __restrict__ v,
                               f16* __restrict__ S_in, f16* __restrict__ S_out) {
    long long tid = (long long)blockIdx.x * blockDim.x + threadIdx.x;
    int e = (int)(tid >> 4);
    if (e >= NE) return;
    int c = (int)(tid & 15) << 2;
    int ue = u[e], ve = v[e];
    float4 hu = *(const float4*)(h + (size_t)ue * DD + c);
    float4 hv = *(const float4*)(h + (size_t)ve * DD + c);
    __half2* si = (__half2*)(S_in  + (size_t)ve * DD + c);
    __half2* so = (__half2*)(S_out + (size_t)ue * DD + c);
    atomicAdd(si + 0, __floats2half2_rn(hu.x, hu.y));
    atomicAdd(si + 1, __floats2half2_rn(hu.z, hu.w));
    atomicAdd(so + 0, __floats2half2_rn(hv.x, hv.y));
    atomicAdd(so + 1, __floats2half2_rn(hv.z, hv.w));
}

// ---- agg[N,128] = X1 @ Wef^T + X2 @ Wer^T + deg_in*bef + deg_out*ber  (f16 out) ----
// X1 = [S_in | deg_in*h | Ein | 0pad], X2 = [deg_out*h | S_out | Eout | 0pad]
__global__ __launch_bounds__(256) void agg_kernel(
    const f16* __restrict__ S_in, const f16* __restrict__ S_out,
    const float* __restrict__ Ein, const float* __restrict__ Eout,
    const float* __restrict__ deg_in, const float* __restrict__ deg_out,
    const float* __restrict__ h,
    const f16* __restrict__ Wef, const f16* __restrict__ Wer,
    const float* __restrict__ bef, const float* __restrict__ ber,
    f16* __restrict__ agg) {
    __shared__ f16 X1[16][MSGP];
    __shared__ f16 X2[16][MSGP];
    int n0 = blockIdx.x * 16;
    int tid = threadIdx.x;
    for (int idx = tid; idx < 16 * MSGP; idx += 256) {
        int m = idx / MSGP, k = idx % MSGP;
        int node = n0 + m;
        f16 x1, x2;
        if (k < 64) {
            x1 = S_in[(size_t)node * DD + k];
            x2 = (f16)(deg_out[node] * h[(size_t)node * DD + k]);
        } else if (k < 128) {
            int j = k - 64;
            x1 = (f16)(deg_in[node] * h[(size_t)node * DD + j]);
            x2 = S_out[(size_t)node * DD + j];
        } else if (k < 144) {
            int j = k - 128;
            x1 = (f16)Ein[(size_t)node * ESZ + j];
            x2 = (f16)Eout[(size_t)node * ESZ + j];
        } else { x1 = (f16)0.0f; x2 = (f16)0.0f; }
        X1[m][k] = x1;
        X2[m][k] = x2;
    }
    __syncthreads();
    int lane = tid & 31, wave = tid >> 5;
    int oBase = wave * 16;
    v8f c = {};
#pragma unroll
    for (int kc = 0; kc < MSGP; kc += 32) {
        v16h a = load_frag_a(&X1[0][0], MSGP, kc, lane);
        v16h b = load_frag_b(Wef, MSGP, oBase, kc, lane);
        c = WMMA_F16(a, b, c);
    }
#pragma unroll
    for (int kc = 0; kc < MSGP; kc += 32) {
        v16h a = load_frag_a(&X2[0][0], MSGP, kc, lane);
        v16h b = load_frag_b(Wer, MSGP, oBase, kc, lane);
        c = WMMA_F16(a, b, c);
    }
    int o  = oBase + (lane & 15);
    int mb = (lane >> 4) << 3;
    float bfo = bef[o], bro = ber[o];
#pragma unroll
    for (int i = 0; i < 8; ++i) {
        int node = n0 + mb + i;
        agg[(size_t)node * HGD + o] = (f16)(c[i] + deg_in[node] * bfo + deg_out[node] * bro);
    }
}

// ---- Fused GRU: gi = agg@W_ih^T+b_ih, gh = h@W_hh^T+b_hh, gate update -> h_out ----
__global__ __launch_bounds__(384) void gru_kernel(
    const f16* __restrict__ agg, const float* __restrict__ h_in,
    const f16* __restrict__ W_ih, const f16* __restrict__ W_hh,
    const float* __restrict__ b_ih, const float* __restrict__ b_hh,
    float* __restrict__ h_out) {
    __shared__ f16  Aa[16][128];
    __shared__ f16  Ah[16][64];
    __shared__ float G[2][16][192];
    int n0 = blockIdx.x * 16;
    int tid = threadIdx.x;
    for (int idx = tid; idx < 16 * 128; idx += 384) {
        int m = idx >> 7, k = idx & 127;
        Aa[m][k] = agg[(size_t)(n0 + m) * HGD + k];
    }
    for (int idx = tid; idx < 16 * 64; idx += 384) {
        int m = idx >> 6, k = idx & 63;
        Ah[m][k] = (f16)h_in[(size_t)(n0 + m) * DD + k];
    }
    __syncthreads();
    int lane = tid & 31, wave = tid >> 5;   // 12 waves -> 12*16 = 192 output cols
    int oBase = wave * 16;
    v8f ci = {}, ch = {};
#pragma unroll
    for (int kc = 0; kc < 128; kc += 32) {
        v16h a = load_frag_a(&Aa[0][0], 128, kc, lane);
        v16h b = load_frag_b(W_ih, 128, oBase, kc, lane);
        ci = WMMA_F16(a, b, ci);
    }
#pragma unroll
    for (int kc = 0; kc < 64; kc += 32) {
        v16h a = load_frag_a(&Ah[0][0], 64, kc, lane);
        v16h b = load_frag_b(W_hh, 64, oBase, kc, lane);
        ch = WMMA_F16(a, b, ch);
    }
    int o  = oBase + (lane & 15);
    int mb = (lane >> 4) << 3;
    float bi = b_ih[o], bh = b_hh[o];
#pragma unroll
    for (int i = 0; i < 8; ++i) {
        G[0][mb + i][o] = ci[i] + bi;
        G[1][mb + i][o] = ch[i] + bh;
    }
    __syncthreads();
    for (int idx = tid; idx < 16 * 64; idx += 384) {
        int m = idx >> 6, j = idx & 63;
        float ir = G[0][m][j],        hr = G[1][m][j];
        float iz = G[0][m][64 + j],   hz = G[1][m][64 + j];
        float inv = G[0][m][128 + j], hn = G[1][m][128 + j];
        float r  = fast_sigmoid(ir + hr);
        float z  = fast_sigmoid(iz + hz);
        float nn = tanhf(inv + r * hn);
        float ho = h_in[(size_t)(n0 + m) * DD + j];
        h_out[(size_t)(n0 + m) * DD + j] = (1.0f - z) * nn + z * ho;
    }
}

// ---- Gated readout: hG[o] = sum_n (h@W_fm^T+b_fm)*sigmoid(h@W_gm^T+b_gm) ----
__global__ __launch_bounds__(256) void readout_kernel(
    const float* __restrict__ h,
    const f16* __restrict__ W_fm, const float* __restrict__ b_fm,
    const f16* __restrict__ W_gm, const float* __restrict__ b_gm,
    float* __restrict__ hG) {
    __shared__ f16 Ah[16][64];
    int n0 = blockIdx.x * 16;
    int tid = threadIdx.x;
    for (int idx = tid; idx < 16 * 64; idx += 256) {
        int m = idx >> 6, k = idx & 63;
        Ah[m][k] = (f16)h[(size_t)(n0 + m) * DD + k];
    }
    __syncthreads();
    int lane = tid & 31, wave = tid >> 5;
    int oBase = wave * 16;
    v8f cf = {}, cg = {};
#pragma unroll
    for (int kc = 0; kc < 64; kc += 32) {
        v16h a  = load_frag_a(&Ah[0][0], 64, kc, lane);
        v16h bf = load_frag_b(W_fm, 64, oBase, kc, lane);
        v16h bg = load_frag_b(W_gm, 64, oBase, kc, lane);
        cf = WMMA_F16(a, bf, cf);
        cg = WMMA_F16(a, bg, cg);
    }
    int o = oBase + (lane & 15);
    float bfo = b_fm[o], bgo = b_gm[o];
    float s = 0.0f;
#pragma unroll
    for (int i = 0; i < 8; ++i)
        s += (cf[i] + bfo) * fast_sigmoid(cg[i] + bgo);
    s += __shfl_xor(s, 16, 32);   // combine M=0..7 (lanes<16) with M=8..15 (lanes>=16)
    if (lane < 16) atomicAdd(&hG[o], s);
}

// ---- Final logit: W_dec . [hG | h[target]] + b_dec ----
__global__ void decision_kernel(const float* __restrict__ hG, const float* __restrict__ h,
                                const int* __restrict__ target,
                                const float* __restrict__ W_dec, const float* __restrict__ b_dec,
                                float* __restrict__ out) {
    __shared__ float red[256];
    int tid = threadIdx.x;
    float s = 0.0f;
    if (tid < HGD + DD) {
        float x = (tid < HGD) ? hG[tid] : h[(size_t)target[0] * DD + (tid - HGD)];
        s = W_dec[tid] * x;
    }
    red[tid] = s;
    __syncthreads();
    for (int off = 128; off > 0; off >>= 1) {
        if (tid < off) red[tid] += red[tid + off];
        __syncthreads();
    }
    if (tid == 0) out[0] = red[0] + b_dec[0];
}

extern "C" void kernel_launch(void* const* d_in, const int* in_sizes, int n_in,
                              void* d_out, int out_size, void* d_ws, size_t ws_size,
                              hipStream_t stream) {
    (void)in_sizes; (void)n_in; (void)out_size; (void)ws_size;
    const float* node_vectors = (const float*)d_in[0];
    const float* edge_vectors = (const float*)d_in[1];
    const float* Wef   = (const float*)d_in[2];
    const float* bef   = (const float*)d_in[3];
    const float* Wer   = (const float*)d_in[4];
    const float* ber   = (const float*)d_in[5];
    const float* W_ih  = (const float*)d_in[6];
    const float* W_hh  = (const float*)d_in[7];
    const float* b_ih  = (const float*)d_in[8];
    const float* b_hh  = (const float*)d_in[9];
    const float* W_fm  = (const float*)d_in[10];
    const float* b_fm  = (const float*)d_in[11];
    const float* W_gm  = (const float*)d_in[12];
    const float* b_gm  = (const float*)d_in[13];
    const float* W_dec = (const float*)d_in[14];
    const float* b_dec = (const float*)d_in[15];
    const int*   u_idx = (const int*)d_in[16];
    const int*   v_idx = (const int*)d_in[17];
    const int*   tgt   = (const int*)d_in[18];

    char* ws = (char*)d_ws;
    size_t off = 0;
    auto salloc = [&](size_t bytes) -> void* {
        void* p = ws + off;
        off = (off + bytes + 255) & ~(size_t)255;
        return p;
    };
    float* Ein   = (float*)salloc((size_t)NN * ESZ * 4);
    float* Eout  = (float*)salloc((size_t)NN * ESZ * 4);
    float* degi  = (float*)salloc((size_t)NN * 4);
    float* dego  = (float*)salloc((size_t)NN * 4);
    f16*   S_in  = (f16*)salloc((size_t)NN * DD * 2);
    f16*   S_out = (f16*)salloc((size_t)NN * DD * 2);
    f16*   agg   = (f16*)salloc((size_t)NN * HGD * 2);
    float* h0    = (float*)salloc((size_t)NN * DD * 4);
    float* h1    = (float*)salloc((size_t)NN * DD * 4);
    float* hG    = (float*)salloc((size_t)HGD * 4);
    f16* Wef16   = (f16*)salloc((size_t)RR * HGD * MSGP * 2);
    f16* Wer16   = (f16*)salloc((size_t)RR * HGD * MSGP * 2);
    f16* Wih16   = (f16*)salloc((size_t)RR * 192 * 128 * 2);
    f16* Whh16   = (f16*)salloc((size_t)RR * 192 * 64 * 2);
    f16* Wfm16   = (f16*)salloc((size_t)HGD * DD * 2);
    f16* Wgm16   = (f16*)salloc((size_t)HGD * DD * 2);

    // One-time weight conversion/padding to f16
    {
        int n1 = RR * HGD * MSGP;   // Wef/Wer padded
        cvt_pad_kernel<<<(n1 + 255) / 256, 256, 0, stream>>>(Wef, Wef16, RR * HGD, MSGK, MSGP);
        cvt_pad_kernel<<<(n1 + 255) / 256, 256, 0, stream>>>(Wer, Wer16, RR * HGD, MSGK, MSGP);
        int n2 = RR * 192 * 128;
        cvt_pad_kernel<<<(n2 + 255) / 256, 256, 0, stream>>>(W_ih, Wih16, RR * 192, 128, 128);
        int n3 = RR * 192 * 64;
        cvt_pad_kernel<<<(n3 + 255) / 256, 256, 0, stream>>>(W_hh, Whh16, RR * 192, 64, 64);
        int n4 = HGD * DD;
        cvt_pad_kernel<<<(n4 + 255) / 256, 256, 0, stream>>>(W_fm, Wfm16, HGD, DD, DD);
        cvt_pad_kernel<<<(n4 + 255) / 256, 256, 0, stream>>>(W_gm, Wgm16, HGD, DD, DD);
    }

    hipMemsetAsync(Ein,  0, (size_t)NN * ESZ * 4, stream);
    hipMemsetAsync(Eout, 0, (size_t)NN * ESZ * 4, stream);
    hipMemsetAsync(degi, 0, (size_t)NN * 4, stream);
    hipMemsetAsync(dego, 0, (size_t)NN * 4, stream);
    hipMemsetAsync(hG,   0, (size_t)HGD * 4, stream);
    hipMemcpyAsync(h0, node_vectors, (size_t)NN * DD * 4, hipMemcpyDeviceToDevice, stream);

    edge_const_kernel<<<(NE + 255) / 256, 256, 0, stream>>>(edge_vectors, u_idx, v_idx,
                                                            Ein, Eout, degi, dego);

    float* hin = h0;
    float* hout = h1;
    for (int r = 0; r < RR; ++r) {
        hipMemsetAsync(S_in,  0, (size_t)NN * DD * 2, stream);
        hipMemsetAsync(S_out, 0, (size_t)NN * DD * 2, stream);
        scatter_kernel<<<(NE * 16) / 256, 256, 0, stream>>>(hin, u_idx, v_idx, S_in, S_out);
        agg_kernel<<<NN / 16, 256, 0, stream>>>(
            S_in, S_out, Ein, Eout, degi, dego, hin,
            Wef16 + (size_t)r * HGD * MSGP, Wer16 + (size_t)r * HGD * MSGP,
            bef + (size_t)r * HGD, ber + (size_t)r * HGD, agg);
        gru_kernel<<<NN / 16, 384, 0, stream>>>(
            agg, hin,
            Wih16 + (size_t)r * 192 * 128, Whh16 + (size_t)r * 192 * 64,
            b_ih + (size_t)r * 192, b_hh + (size_t)r * 192, hout);
        float* t = hin; hin = hout; hout = t;
    }

    readout_kernel<<<NN / 16, 256, 0, stream>>>(hin, Wfm16, b_fm, Wgm16, b_gm, hG);
    decision_kernel<<<1, 256, 0, stream>>>(hG, hin, tgt, W_dec, b_dec, (float*)d_out);
}
